// graphSAGEmodel_54863912239768
// MI455X (gfx1250) — matrix operand order repaired and checked by reading, not compiled, not run on hardware
//
#include <hip/hip_runtime.h>
#include <stdint.h>

// ---------- CDNA5 vector types ----------
typedef __bf16        v16bf  __attribute__((ext_vector_type(16)));
typedef __bf16        v2bf   __attribute__((ext_vector_type(2)));
typedef float         v8f    __attribute__((ext_vector_type(8)));
typedef unsigned int  u32x4  __attribute__((ext_vector_type(4)));
typedef unsigned int  u32x2  __attribute__((ext_vector_type(2)));
typedef int           i32x4  __attribute__((ext_vector_type(4)));

#if defined(__has_builtin)
#if __has_builtin(__builtin_amdgcn_global_load_async_to_lds_b128)
#define SAGE_ASYNC 1
#endif
#endif
#ifndef SAGE_ASYNC
#define SAGE_ASYNC 0
#endif

// builtin signature (from clang diagnostic): (int4 AS1*, int4 AS3*, imm offset, imm cpol)
#define SAGE_GAS(p) ((__attribute__((address_space(1))) i32x4*)(void*)(p))
#define SAGE_LAS(p) ((__attribute__((address_space(3))) i32x4*)(void*)(p))

__device__ __forceinline__ unsigned short sage_f2bf(float f) {
  unsigned int u = __float_as_uint(f);
  u += 0x7fffu + ((u >> 16) & 1u);   // RNE
  return (unsigned short)(u >> 16);
}

__device__ __forceinline__ unsigned sage_pk_bf16(float lo, float hi) {
#if defined(__has_builtin)
#if __has_builtin(__builtin_amdgcn_cvt_pk_bf16_f32)
  union { v2bf v; unsigned u; } c;
  c.v = __builtin_amdgcn_cvt_pk_bf16_f32(lo, hi);
  return c.u;
#else
  return (unsigned)sage_f2bf(lo) | ((unsigned)sage_f2bf(hi) << 16);
#endif
#else
  return (unsigned)sage_f2bf(lo) | ((unsigned)sage_f2bf(hi) << 16);
#endif
}

__device__ __forceinline__ float sage_bf2f(unsigned short h) {
  return __uint_as_float((unsigned)h << 16);
}

__device__ __forceinline__ void sage_wait_async_le3() {
#if defined(__has_builtin) && __has_builtin(__builtin_amdgcn_s_wait_asynccnt)
  __builtin_amdgcn_s_wait_asynccnt(3);
#elif SAGE_ASYNC
  asm volatile("s_wait_asynccnt 0x3" ::: "memory");
#endif
}
__device__ __forceinline__ void sage_wait_async_le0() {
#if defined(__has_builtin) && __has_builtin(__builtin_amdgcn_s_wait_asynccnt)
  __builtin_amdgcn_s_wait_asynccnt(0);
#elif SAGE_ASYNC
  asm volatile("s_wait_asynccnt 0x0" ::: "memory");
#endif
}

// ---------- degree / normalization ----------
__global__ __launch_bounds__(256) void sage_deg(const int* __restrict__ dst,
                                                float* __restrict__ deg, int nE) {
  int e = blockIdx.x * blockDim.x + threadIdx.x;
  if (e < nE) atomicAdd(&deg[dst[e]], 1.0f);
}

__global__ __launch_bounds__(256) void sage_inv(const float* __restrict__ deg,
                                                float* __restrict__ inv, int n) {
  int i = blockIdx.x * blockDim.x + threadIdx.x;
  if (i < n) inv[i] = 1.0f / fmaxf(deg[i], 1.0f);
}

// ---------- f32 -> bf16 convert (optionally row-scaled): pairs of elements ----------
__global__ __launch_bounds__(256) void sage_cvt(const float* __restrict__ src,
                                                unsigned short* __restrict__ dst,
                                                const float* __restrict__ inv,
                                                long long nPairs, int rowShift) {
  long long t = (long long)blockIdx.x * blockDim.x + threadIdx.x;
  if (t >= nPairs) return;
  float2 v = ((const float2*)src)[t];
  float s = inv ? inv[t >> rowShift] : 1.0f;
  ((unsigned*)dst)[t] = sage_pk_bf16(v.x * s, v.y * s);
}

// ---------- edge scatter-add: agg[dst] += feat_bf16[src]  (d = 4 << chunkShift) ----------
__global__ __launch_bounds__(256) void sage_scatter(const unsigned short* __restrict__ feat,
                                                    const int* __restrict__ src,
                                                    const int* __restrict__ dst,
                                                    float* __restrict__ agg,
                                                    int nE, int chunkShift) {
  const int chunks = 1 << chunkShift;
  const int d = chunks << 2;
  long long t = (long long)blockIdx.x * blockDim.x + threadIdx.x;
  long long total = (long long)nE << chunkShift;
  if (t >= total) return;
  int e = (int)(t >> chunkShift);
  int c = (int)(t & (chunks - 1)) << 2;
  int s = src[e], q = dst[e];
  u32x2 pv = *(const u32x2*)(feat + (size_t)s * d + c);
  float* o = agg + (size_t)q * d + c;
  atomicAdd(o + 0, sage_bf2f((unsigned short)(pv.x & 0xffffu)));
  atomicAdd(o + 1, sage_bf2f((unsigned short)(pv.x >> 16)));
  atomicAdd(o + 2, sage_bf2f((unsigned short)(pv.y & 0xffffu)));
  atomicAdd(o + 3, sage_bf2f((unsigned short)(pv.y >> 16)));
}

// ---------- weight convert + transpose: Wt[n][kOff+k] = bf16(W[k][n]) ----------
__global__ __launch_bounds__(256) void sage_wt(const float* __restrict__ W,
                                               unsigned short* __restrict__ Wt,
                                               int K, int N, int nRows,
                                               int pitchK, int kOff) {
  int idx = blockIdx.x * blockDim.x + threadIdx.x;
  if (idx >= nRows * K) return;
  int n = idx / K, k = idx - n * K;
  float v = (n < N) ? W[(size_t)k * N + n] : 0.0f;
  Wt[(size_t)n * pitchK + kOff + k] = sage_f2bf(v);
}

// ---------- fused SAGE GEMM (pure bf16 operands, async-LDS double-buffered) ----------
// C = act( [A0|A1] @ Wt^T + bias ),  A0: M x K0 bf16, A1: M x K1 bf16 (rows padded!)
// Wt: [nRows][K0+K1] bf16 N-major.  Output f32 or bf16.
#define BM 128
#define BN 64
#define BK 32
#define LP 40   // LDS ushort pitch (80B rows -> conflict-free ds_load_b128)

union SageFrag { struct { u32x4 a, b; } q; v16bf v; };

__global__ __launch_bounds__(256)
void sage_gemm(const unsigned short* __restrict__ A0,
               const unsigned short* __restrict__ A1,
               int K0, int K1,
               const unsigned short* __restrict__ Wt, int ldWt,
               const float* __restrict__ bias,
               void* __restrict__ Cout, int M, int Nout, int ldC,
               int relu, int outBf16) {
  __shared__ __align__(16) unsigned short ldsA[2][BM * LP];
  __shared__ __align__(16) unsigned short ldsB[2][BN * LP];

  const int tid  = threadIdx.x;
  const int lane = tid & 31;
  const int wave = tid >> 5;
  const int wm   = wave & 3;         // 4 waves along M
  const int wn   = wave >> 2;        // 2 waves along N
  const int l15  = lane & 15;
  const int hf   = lane >> 4;        // K-half select (contiguous k = 16*hf + s, same in A and B)
  const int mBase = blockIdx.x * BM;
  const int nBase = blockIdx.y * BN;

  v8f acc00 = {}, acc01 = {}, acc10 = {}, acc11 = {};
  const int nSteps = (K0 + K1) / BK;

#if SAGE_ASYNC
  // issue one tile's loads: 2x b128 per thread for A (128x32 bf16), 1x for B (64x32 bf16)
  auto issue_tile = [&](int step, int buf) {
    int k0 = step * BK;
    const unsigned short* Aseg; int lda, kLoc;
    if (k0 < K0) { Aseg = A0; lda = K0; kLoc = k0; }
    else         { Aseg = A1; lda = K1; kLoc = k0 - K0; }
#pragma unroll
    for (int j = 0; j < 2; ++j) {
      int id = tid + 256 * j;                  // 0..511 16B slots
      int r = id >> 2, c8 = (id & 3) << 3;     // row 0..127, col 0/8/16/24
      const unsigned short* g = Aseg + (size_t)(mBase + r) * lda + kLoc + c8;
      __builtin_amdgcn_global_load_async_to_lds_b128(
          SAGE_GAS(g), SAGE_LAS(&ldsA[buf][r * LP + c8]), 0, 0);
    }
    {
      int r = tid >> 2, c8 = (tid & 3) << 3;   // row 0..63
      const unsigned short* g = Wt + (size_t)(nBase + r) * ldWt + k0 + c8;
      __builtin_amdgcn_global_load_async_to_lds_b128(
          SAGE_GAS(g), SAGE_LAS(&ldsB[buf][r * LP + c8]), 0, 0);
    }
  };
  issue_tile(0, 0);
#endif

  for (int s = 0; s < nSteps; ++s) {
#if SAGE_ASYNC
    const int cur = s & 1;
    if (s + 1 < nSteps) { issue_tile(s + 1, cur ^ 1); sage_wait_async_le3(); }
    else                { sage_wait_async_le0(); }
#else
    const int cur = 0;
    {   // synchronous tile copy (fallback)
      int k0 = s * BK;
      const unsigned short* Aseg; int lda, kLoc;
      if (k0 < K0) { Aseg = A0; lda = K0; kLoc = k0; }
      else         { Aseg = A1; lda = K1; kLoc = k0 - K0; }
#pragma unroll
      for (int j = 0; j < 2; ++j) {
        int id = tid + 256 * j;
        int r = id >> 2, c8 = (id & 3) << 3;
        u32x4 v = *(const u32x4*)(Aseg + (size_t)(mBase + r) * lda + kLoc + c8);
        *(u32x4*)&ldsA[0][r * LP + c8] = v;
      }
      {
        int r = tid >> 2, c8 = (tid & 3) << 3;
        u32x4 v = *(const u32x4*)(Wt + (size_t)(nBase + r) * ldWt + k0 + c8);
        *(u32x4*)&ldsB[0][r * LP + c8] = v;
      }
    }
#endif
    __syncthreads();

    // --- fragments (A: lane=row, contiguous k = 16*hf + s; B: N-major, same k order) ---
    SageFrag a0, a1, b0, b1;
    const unsigned short* lA = &ldsA[cur][0];
    const unsigned short* lB = &ldsB[cur][0];
    const int ar0 = (wm * 32 + l15) * LP;
    const int ar1 = (wm * 32 + 16 + l15) * LP;
    const int ka0 = hf * 16, ka1 = hf * 16 + 8;
    a0.q.a = *(const u32x4*)&lA[ar0 + ka0];
    a0.q.b = *(const u32x4*)&lA[ar0 + ka1];
    a1.q.a = *(const u32x4*)&lA[ar1 + ka0];
    a1.q.b = *(const u32x4*)&lA[ar1 + ka1];
    const int br0 = (wn * 32 + l15) * LP + hf * 16;
    const int br1 = (wn * 32 + 16 + l15) * LP + hf * 16;
    b0.q.a = *(const u32x4*)&lB[br0];
    b0.q.b = *(const u32x4*)&lB[br0 + 8];
    b1.q.a = *(const u32x4*)&lB[br1];
    b1.q.b = *(const u32x4*)&lB[br1 + 8];

    acc00 = __builtin_amdgcn_wmma_f32_16x16x32_bf16(false, a0.v, false, b0.v,
                                                    (short)0, acc00, false, false);
    acc01 = __builtin_amdgcn_wmma_f32_16x16x32_bf16(false, a0.v, false, b1.v,
                                                    (short)0, acc01, false, false);
    acc10 = __builtin_amdgcn_wmma_f32_16x16x32_bf16(false, a1.v, false, b0.v,
                                                    (short)0, acc10, false, false);
    acc11 = __builtin_amdgcn_wmma_f32_16x16x32_bf16(false, a1.v, false, b1.v,
                                                    (short)0, acc11, false, false);
    __syncthreads();
  }

  // --- epilogue: bias + ReLU, f32 or bf16 stores (C/D layout: row = r + 8*hf) ---
  auto store_tile = [&](v8f acc, int mOff, int nOff) {
    int col = nBase + wn * 32 + nOff + l15;
    if (col >= Nout) return;
    float bj = bias[col];
    int rowB = mBase + wm * 32 + mOff + hf * 8;
#pragma unroll
    for (int r = 0; r < 8; ++r) {
      int row = rowB + r;
      if (row < M) {
        float v = acc[r] + bj;
        if (relu) v = fmaxf(v, 0.0f);
        if (outBf16) ((unsigned short*)Cout)[(size_t)row * ldC + col] = sage_f2bf(v);
        else         ((float*)Cout)[(size_t)row * ldC + col] = v;
      }
    }
  };
  store_tile(acc00, 0, 0);  store_tile(acc01, 0, 16);
  store_tile(acc10, 16, 0); store_tile(acc11, 16, 16);
}

// ---------------- host launcher ----------------
extern "C" void kernel_launch(void* const* d_in, const int* in_sizes, int n_in,
                              void* d_out, int out_size, void* d_ws, size_t ws_size,
                              hipStream_t stream) {
  const float* x   = (const float*)d_in[0];
  const int*   ei  = (const int*)d_in[1];
  const float* W1s = (const float*)d_in[2];
  const float* W1n = (const float*)d_in[3];
  const float* b1  = (const float*)d_in[4];
  const float* W2s = (const float*)d_in[5];
  const float* W2n = (const float*)d_in[6];
  const float* b2  = (const float*)d_in[7];
  const float* Wh  = (const float*)d_in[8];
  const float* bh  = (const float*)d_in[9];
  float* out = (float*)d_out;

  const int N = in_sizes[0] / 128;      // 100000
  const int E = in_sizes[1] / 2;        // 500000
  const int DIN = 128, H = 256, OUTC = 349;
  const int NP = N + BM;                // pad rows: tile loads need no M-guard
  const int* src = ei;
  const int* dst = ei + E;

  char* ws = (char*)d_ws;
  size_t off = 0;
  auto give = [&](size_t bytes) -> char* {
    char* p = ws + off;
    off = (off + bytes + 255) & ~(size_t)255;
    return p;
  };
  float* deg  = (float*)give((size_t)N * 4);
  float* inv  = (float*)give((size_t)N * 4);
  float* agg1 = (float*)give((size_t)N * DIN * 4);          // f32 atomic targets
  float* agg2 = (float*)give((size_t)N * H * 4);
  unsigned short* xb    = (unsigned short*)give((size_t)NP * DIN * 2);  // bf16 features
  unsigned short* agg1b = (unsigned short*)give((size_t)NP * DIN * 2);
  unsigned short* agg2b = (unsigned short*)give((size_t)NP * H * 2);
  unsigned short* h1b   = (unsigned short*)give((size_t)NP * H * 2);
  unsigned short* h2b   = (unsigned short*)give((size_t)NP * H * 2);
  unsigned short* Wt1 = (unsigned short*)give((size_t)256 * 256 * 2);
  unsigned short* Wt2 = (unsigned short*)give((size_t)256 * 512 * 2);
  unsigned short* Wth = (unsigned short*)give((size_t)384 * 256 * 2);

  (void)hipMemsetAsync(deg,  0, (size_t)N * 4, stream);
  (void)hipMemsetAsync(agg1, 0, (size_t)N * DIN * 4, stream);
  (void)hipMemsetAsync(agg2, 0, (size_t)N * H * 4, stream);

  sage_deg<<<(E + 255) / 256, 256, 0, stream>>>(dst, deg, E);
  sage_inv<<<(N + 255) / 256, 256, 0, stream>>>(deg, inv, N);

  // x -> bf16 once (halves gather + GEMM read traffic)
  long long px = (long long)N * (DIN / 2);
  sage_cvt<<<(unsigned)((px + 255) / 256), 256, 0, stream>>>(x, xb, nullptr, px, 6);

  // weights: bf16, transposed, K-concatenated [W_neigh ; W_self]
  sage_wt<<<(256 * 128 + 255) / 256, 256, 0, stream>>>(W1n, Wt1, 128, 256, 256, 256, 0);
  sage_wt<<<(256 * 128 + 255) / 256, 256, 0, stream>>>(W1s, Wt1, 128, 256, 256, 256, 128);
  sage_wt<<<(256 * 256 + 255) / 256, 256, 0, stream>>>(W2n, Wt2, 256, 256, 256, 512, 0);
  sage_wt<<<(256 * 256 + 255) / 256, 256, 0, stream>>>(W2s, Wt2, 256, 256, 256, 512, 256);
  sage_wt<<<(384 * 256 + 255) / 256, 256, 0, stream>>>(Wh,  Wth, 256, 349, 384, 256, 0);

  const int gx = (N + BM - 1) / BM;     // 782 M-tiles

  // ---- layer 1 ----
  long long t1 = (long long)E * (DIN / 4);
  sage_scatter<<<(unsigned)((t1 + 255) / 256), 256, 0, stream>>>(xb, src, dst, agg1, E, 5);
  sage_cvt<<<(unsigned)((px + 255) / 256), 256, 0, stream>>>(agg1, agg1b, inv, px, 6);
  sage_gemm<<<dim3(gx, 4), 256, 0, stream>>>(agg1b, xb, 128, 128, Wt1, 256, b1,
                                             h1b, N, 256, 256, 1, 1);

  // ---- layer 2 ----
  long long t2 = (long long)E * (H / 4);
  sage_scatter<<<(unsigned)((t2 + 255) / 256), 256, 0, stream>>>(h1b, src, dst, agg2, E, 6);
  long long p2 = (long long)N * (H / 2);
  sage_cvt<<<(unsigned)((p2 + 255) / 256), 256, 0, stream>>>(agg2, agg2b, inv, p2, 7);
  sage_gemm<<<dim3(gx, 4), 256, 0, stream>>>(agg2b, h1b, 256, 256, Wt2, 512, b2,
                                             h2b, N, 256, 256, 1, 1);

  // ---- head (no ReLU), f32 out, N=349 over 6x64 tiles ----
  sage_gemm<<<dim3(gx, 6), 256, 0, stream>>>(h2b, h2b, 256, 0, Wth, 256, bh,
                                             out, N, OUTC, OUTC, 0, 0);

  (void)n_in; (void)out_size; (void)ws_size;
}
